// rev_vit_snn_21655225106979
// MI455X (gfx1250) — 16-bit and 8-bit matrix operand order repaired and checked by reading, compiled, not run on hardware
//
#include <hip/hip_runtime.h>

typedef __attribute__((ext_vector_type(16))) _Float16 v16h;
typedef __attribute__((ext_vector_type(8)))  _Float16 v8h;
typedef __attribute__((ext_vector_type(8)))  float    v8f;
typedef __attribute__((ext_vector_type(8)))  int      v8i;

#define T_STEPS 4
#define BATCH   2
#define TBN     8
#define EMB     256
#define NTOK    1024
#define NHEADS  8

__device__ __forceinline__ int a16_k(int j, int grp) { return (j < 8 ? j : j + 8) + grp * 8; }
__device__ __forceinline__ int b16_k(int j, int grp) { return j + grp * 16; }

__device__ __forceinline__ v16h cat16(v8h lo, v8h hi) {
  v16h r;
#pragma unroll
  for (int j = 0; j < 8; ++j) { r[j] = lo[j]; r[j + 8] = hi[j]; }
  return r;
}
__device__ __forceinline__ v8f wmma_f16(v16h a, v16h b, v8f c) {
  return __builtin_amdgcn_wmma_f32_16x16x32_f16(false, a, false, b, (short)0, c, false, false);
}
__device__ __forceinline__ v8i wmma_u8(v8i a, v8i b, v8i c) {
  return __builtin_amdgcn_wmma_i32_16x16x64_iu8(false, a, false, b, c, false, false);
}

__global__ void pack_f16_kernel(const float* __restrict__ w, _Float16* __restrict__ o, int n) {
  int i = blockIdx.x * blockDim.x + threadIdx.x;
  if (i < n) o[i] = (_Float16)w[i];
}

__global__ void pack_conv_kernel(const float* __restrict__ w, _Float16* __restrict__ o,
                                 int I, int n) {
  int i = blockIdx.x * blockDim.x + threadIdx.x;
  if (i >= n) return;
  int r  = i % 9;
  int ch = (i / 9) % I;
  int oc = i / (9 * I);
  o[(long)oc * I * 9 + (ch >> 5) * 288 + r * 32 + (ch & 31)] = (_Float16)w[i];
}

__global__ void bn_prep_kernel(const float* __restrict__ g, const float* __restrict__ b,
                               const float* __restrict__ m, const float* __restrict__ v,
                               const float* __restrict__ bias,
                               float* __restrict__ es, float* __restrict__ eh, int C) {
  int i = blockIdx.x * blockDim.x + threadIdx.x;
  if (i >= C) return;
  float s = g[i] * rsqrtf(v[i] + 1e-5f);
  float h = b[i] - m[i] * s;
  if (bias) h += bias[i] * s;
  es[i] = s; eh[i] = h;
}

__global__ void conv0_kernel(const float* __restrict__ X, const float* __restrict__ W0,
                             const float* __restrict__ es, const float* __restrict__ eh,
                             float* __restrict__ A0) {
  long i = (long)blockIdx.x * blockDim.x + threadIdx.x;
  if (i >= (long)BATCH * 32 * 128 * 128) return;
  int x = i & 127; long r = i >> 7;
  int y = r & 127; r >>= 7;
  int c = r & 31;  int b = (int)(r >> 5);
  float acc = 0.f;
  for (int ci = 0; ci < 2; ++ci)
    for (int ky = 0; ky < 3; ++ky) {
      int gy = y - 1 + ky; if (gy < 0 || gy > 127) continue;
      for (int kx = 0; kx < 3; ++kx) {
        int gx = x - 1 + kx; if (gx < 0 || gx > 127) continue;
        acc += X[(((long)b * 2 + ci) * 128 + gy) * 128 + gx] * W0[((c * 2 + ci) * 3 + ky) * 3 + kx];
      }
    }
  A0[i] = acc * es[c] + eh[c];
}

__global__ void lif_kernel(const float* __restrict__ X, _Float16* __restrict__ Sf,
                           _Float16* __restrict__ Sft, unsigned char* __restrict__ Su,
                           float vth, long S, long xStride, int C, int NT) {
  long i = (long)blockIdx.x * blockDim.x + threadIdx.x;
  if (i >= S) return;
  long ti = 0;
  if (Sft) {
    int n = (int)(i % NT);
    int c = (int)((i / NT) % C);
    long b = i / ((long)NT * C);
    ti = (b * NT + n) * C + c;
  }
  float v = 0.f;
  for (int t = 0; t < T_STEPS; ++t) {
    float x = X[t * xStride + i];
    v += (x - v) * 0.5f;
    float sp = (v >= vth) ? 1.f : 0.f;
    if (Sf)  Sf[t * S + i]  = (_Float16)sp;
    if (Sft) Sft[t * S + ti] = (_Float16)sp;
    if (Su)  Su[t * S + i]  = (unsigned char)sp;
    v *= (1.f - sp);
  }
}

__global__ void maxpool_kernel(const _Float16* __restrict__ In, _Float16* __restrict__ Out,
                               int C, int H, int W, long total) {
  long i = (long)blockIdx.x * blockDim.x + threadIdx.x;
  if (i >= total) return;
  int Ho = H >> 1, Wo = W >> 1;
  int xo = (int)(i % Wo); long r = i / Wo;
  int yo = (int)(r % Ho); r /= Ho;
  float m = 0.f;
  for (int ky = 0; ky < 3; ++ky) {
    int y = yo * 2 - 1 + ky; if (y < 0 || y >= H) continue;
    for (int kx = 0; kx < 3; ++kx) {
      int x = xo * 2 - 1 + kx; if (x < 0 || x >= W) continue;
      m = fmaxf(m, (float)In[(r * H + y) * W + x]);
    }
  }
  Out[i] = (_Float16)m;
}

__global__ void __launch_bounds__(128)
conv3x3_wmma_kernel(const _Float16* __restrict__ S, const _Float16* __restrict__ Wp,
                    float* __restrict__ D, const float* __restrict__ es,
                    const float* __restrict__ eh, int I, int O, int H, int W) {
  __shared__ __align__(32) _Float16 patch[3 * 18 * 32];
  const int tid = threadIdx.x;
  const int lane = tid & 31, wave = tid >> 5;
  const int grp = lane >> 4, l16 = lane & 15;
  const int wtiles = W >> 4;
  const int y  = blockIdx.x / wtiles;
  const int x0 = (blockIdx.x % wtiles) << 4;
  const int mt = blockIdx.y * 64 + wave * 16;
  const int tb = blockIdx.z;
  const int Ktot = I * 9;
  const long inb = (long)tb * I * H * W;
  v8f acc = {};
  for (int i0 = 0; i0 < I; i0 += 32) {
    __syncthreads();
    for (int e = tid; e < 3 * 18 * 32; e += 128) {
      int c = e & 31, rc = e >> 5;
      int col = rc % 18, row = rc / 18;
      int gy = y - 1 + row, gx = x0 - 1 + col;
      _Float16 v = (_Float16)0.f;
      if (gy >= 0 && gy < H && gx >= 0 && gx < W)
        v = S[inb + (long)(i0 + c) * H * W + (long)gy * W + gx];
      patch[e] = v;
    }
    __syncthreads();
    const _Float16* wrow = Wp + (long)(mt + l16) * Ktot + (long)i0 * 9;
#pragma unroll
    for (int r = 0; r < 9; ++r) {
      const int ky = r / 3, kx = r % 3;
      v8h alo = *(const v8h*)(wrow + r * 32 + grp * 8);
      v8h ahi = *(const v8h*)(wrow + r * 32 + grp * 8 + 16);
      const v8h* pb = (const v8h*)(&patch[(ky * 18 + l16 + kx) * 32 + grp * 8]);
      acc = wmma_f16(cat16(alo, ahi), cat16(pb[0], pb[2]), acc);
    }
  }
#pragma unroll
  for (int r = 0; r < 8; ++r) {
    int m = mt + r + grp * 8;
    D[(long)tb * O * H * W + (long)m * H * W + (long)y * W + x0 + l16] = acc[r] * es[m] + eh[m];
  }
}

__global__ void __launch_bounds__(128)
gemm_wmma_kernel(const _Float16* __restrict__ A, const _Float16* __restrict__ Bt,
                 float* __restrict__ D, const float* __restrict__ es,
                 const float* __restrict__ eh, const float* __restrict__ resid,
                 int M, int N, int K) {
  __shared__ __align__(32) _Float16 lb[16 * 32];
  const int tid = threadIdx.x;
  const int lane = tid & 31, wave = tid >> 5;
  const int grp = lane >> 4, l16 = lane & 15;
  const int nt = blockIdx.x * 16;
  const int mt = blockIdx.y * 64 + wave * 16;
  const _Float16* Bp = Bt + (long)blockIdx.z * N * K;
  const long db = (long)blockIdx.z * M * N;
  const _Float16* Arow = A + (long)(mt + l16) * K;
  const int n_l = tid >> 3, kq = (tid & 7) * 4;
  v8f acc = {};
  for (int k0 = 0; k0 < K; k0 += 32) {
    __syncthreads();
    const _Float16* g = Bp + (long)(nt + n_l) * K + k0 + kq;
    unsigned ldsa = (unsigned)(unsigned long long)&lb[n_l * 32 + kq];
    asm volatile("global_load_async_to_lds_b64 %0, %1, off" :: "v"(ldsa), "v"(g) : "memory");
    if (k0 + 32 < K) __builtin_prefetch(g + 32, 0, 1);
    asm volatile("s_wait_asynccnt 0x0" ::: "memory");
    __syncthreads();
    v8h alo = *(const v8h*)(Arow + k0 + grp * 8);
    v8h ahi = *(const v8h*)(Arow + k0 + grp * 8 + 16);
    const v8h* pb = (const v8h*)(&lb[l16 * 32 + grp * 8]);
    acc = wmma_f16(cat16(alo, ahi), cat16(pb[0], pb[2]), acc);
  }
#pragma unroll
  for (int r = 0; r < 8; ++r) {
    int m = mt + r + grp * 8;
    long idx = db + (long)m * N + nt + l16;
    float v = acc[r] * es[m] + eh[m];
    if (resid) v += resid[idx];
    D[idx] = v;
  }
}

__global__ void __launch_bounds__(32)
kTv_iu8_kernel(const unsigned char* __restrict__ Ksp, const unsigned char* __restrict__ Vsp,
               _Float16* __restrict__ KTVt) {
  const int lane = threadIdx.x & 31;
  const int grp = lane >> 4, l16 = lane & 15;
  const int h = blockIdx.x % NHEADS, tb = blockIdx.x / NHEADS;
  const unsigned char* kb = Ksp + ((long)tb * EMB + h * 32) * NTOK;
  const unsigned char* vb = Vsp + ((long)tb * EMB + h * 32) * NTOK;
  v8i a00 = {}, a01 = {}, a10 = {}, a11 = {};
  for (int k0 = 0; k0 < NTOK; k0 += 64) {
    v8i fa0, fa1, fb0, fb1;
#pragma unroll
    for (int vg = 0; vg < 8; ++vg) {
      int ab = (vg & 1) * 4 + ((vg >> 1) & 1) * 16 + ((vg >> 2) & 1) * 32 + grp * 8 + k0;
      fa0[vg] = *(const int*)(kb + (long)l16 * NTOK + ab);
      fa1[vg] = *(const int*)(kb + (long)(16 + l16) * NTOK + ab);
      int bb = (vg & 1) * 4 + ((vg >> 1) & 1) * 16 + ((vg >> 2) & 1) * 32 + grp * 8 + k0;
      fb0[vg] = *(const int*)(vb + (long)l16 * NTOK + bb);
      fb1[vg] = *(const int*)(vb + (long)(16 + l16) * NTOK + bb);
    }
    a00 = wmma_u8(fa0, fb0, a00); a01 = wmma_u8(fa0, fb1, a01);
    a10 = wmma_u8(fa1, fb0, a10); a11 = wmma_u8(fa1, fb1, a11);
  }
  _Float16* out = KTVt + (long)blockIdx.x * 32 * 32;
#pragma unroll
  for (int r = 0; r < 8; ++r) {
    int m = r + grp * 8;
    out[l16 * 32 + m]              = (_Float16)(float)a00[r];
    out[(16 + l16) * 32 + m]       = (_Float16)(float)a01[r];
    out[l16 * 32 + 16 + m]         = (_Float16)(float)a10[r];
    out[(16 + l16) * 32 + 16 + m]  = (_Float16)(float)a11[r];
  }
}

__global__ void __launch_bounds__(128)
qkTv_kernel(const _Float16* __restrict__ Qt, const _Float16* __restrict__ KTVt,
            float* __restrict__ Oc) {
  const int tid = threadIdx.x, lane = tid & 31, wave = tid >> 5;
  const int grp = lane >> 4, l16 = lane & 15;
  const int tok0 = (blockIdx.x * 4 + wave) * 16;
  const int h = blockIdx.y, tb = blockIdx.z;
  const _Float16* qrow = Qt + ((long)tb * NTOK + tok0 + l16) * EMB + h * 32;
  v8h alo = *(const v8h*)(qrow + grp * 8);
  v8h ahi = *(const v8h*)(qrow + grp * 8 + 16);
  v16h af = cat16(alo, ahi);
  const _Float16* kt = KTVt + ((long)tb * NHEADS + h) * 1024;
  const v8h* p0 = (const v8h*)(kt + (long)l16 * 32 + grp * 8);
  const v8h* p1 = (const v8h*)(kt + (long)(16 + l16) * 32 + grp * 8);
  v8f c0 = {}, c1 = {};
  c0 = wmma_f16(af, cat16(p0[0], p0[2]), c0);
  c1 = wmma_f16(af, cat16(p1[0], p1[2]), c1);
  float* ob = Oc + ((long)tb * EMB + h * 32) * NTOK;
#pragma unroll
  for (int r = 0; r < 8; ++r) {
    int m = tok0 + r + grp * 8;
    ob[(long)l16 * NTOK + m]        = c0[r] * 0.125f;
    ob[(long)(16 + l16) * NTOK + m] = c1[r] * 0.125f;
  }
}

__global__ void feat_kernel(const float* __restrict__ x1, const float* __restrict__ x2,
                            float* __restrict__ feat) {
  int i = blockIdx.x * blockDim.x + threadIdx.x;
  if (i >= BATCH * 512) return;
  int c2 = i % 512, b = i / 512;
  const float* src = (c2 < EMB) ? x1 : x2;
  int c = c2 & (EMB - 1);
  float s = 0.f;
  for (int t = 0; t < T_STEPS; ++t) {
    const float* p = src + ((long)(t * BATCH + b) * EMB + c) * NTOK;
    for (int n = 0; n < NTOK; ++n) s += p[n];
  }
  feat[i] = s * (1.f / (T_STEPS * NTOK));
}

__global__ void head_kernel(const float* __restrict__ feat, const float* __restrict__ hw,
                            const float* __restrict__ hb, float* __restrict__ out) {
  int i = blockIdx.x * blockDim.x + threadIdx.x;
  if (i >= BATCH * 1000) return;
  int cls = i % 1000, b = i / 1000;
  float s = hb[cls];
  for (int c = 0; c < 512; ++c) s += feat[b * 512 + c] * hw[(long)cls * 512 + c];
  out[i] = s;
}

extern "C" void kernel_launch(void* const* d_in, const int* in_sizes, int n_in,
                              void* d_out, int out_size, void* d_ws, size_t ws_size,
                              hipStream_t stream) {
  (void)in_sizes; (void)out_size;
  if (n_in < 94) return;
  auto F = [&](int i) { return (const float*)d_in[i]; };
  const float* hb = F(0);
  const float* hw = F(1);
  const float* xin = F(93);

  size_t cur = 0;
  auto alloc = [&](size_t bytes) -> char* {
    cur = (cur + 255) & ~(size_t)255;
    char* p = (char*)d_ws + cur; cur += bytes; return p;
  };
  float*    a0  = (float*)   alloc((size_t)BATCH * 32 * 16384 * 4);
  _Float16* s0  = (_Float16*)alloc((size_t)TBN * 32 * 16384 * 2);
  float*    a1  = (float*)   alloc((size_t)TBN * 64 * 16384 * 4);
  _Float16* s1  = (_Float16*)alloc((size_t)TBN * 64 * 16384 * 2);
  float*    a2  = (float*)   alloc((size_t)TBN * 128 * 16384 * 4);
  _Float16* s2  = (_Float16*)alloc((size_t)TBN * 128 * 16384 * 2);
  _Float16* s2p = (_Float16*)alloc((size_t)TBN * 128 * 4096 * 2);
  float*    a3  = (float*)   alloc((size_t)TBN * 256 * 4096 * 4);
  _Float16* s3  = (_Float16*)alloc((size_t)TBN * 256 * 4096 * 2);
  _Float16* s3p = (_Float16*)alloc((size_t)TBN * 256 * 1024 * 2);
  size_t scratch_limit = cur;
  float*    x1  = (float*)   alloc((size_t)TBN * EMB * NTOK * 4);
  float*    x2  = (float*)   alloc((size_t)TBN * EMB * NTOK * 4);
  _Float16* w1p = (_Float16*)alloc((size_t)64 * 288 * 2);
  _Float16* w2p = (_Float16*)alloc((size_t)128 * 576 * 2);
  _Float16* w3p = (_Float16*)alloc((size_t)256 * 1152 * 2);
  _Float16* w4p = (_Float16*)alloc((size_t)256 * 2304 * 2);
  _Float16 *qwp[2], *kwp[2], *vwp[2], *pwp[2], *m1p[2], *m2p[2];
  for (int L = 0; L < 2; ++L) {
    qwp[L] = (_Float16*)alloc(65536 * 2);  kwp[L] = (_Float16*)alloc(65536 * 2);
    vwp[L] = (_Float16*)alloc(65536 * 2);  pwp[L] = (_Float16*)alloc(65536 * 2);
    m1p[L] = (_Float16*)alloc(262144 * 2); m2p[L] = (_Float16*)alloc(262144 * 2);
  }
  float *esA[17], *ehA[17];
  for (int i = 0; i < 17; ++i) { esA[i] = (float*)alloc(1024 * 4); ehA[i] = (float*)alloc(1024 * 4); }
  size_t need = cur;

  size_t scur = 0;
  auto sal = [&](size_t bytes) -> char* {
    scur = (scur + 255) & ~(size_t)255;
    char* p = (char*)d_ws + scur; scur += bytes; return p;
  };
  _Float16* st   = (_Float16*)sal((size_t)TBN * NTOK * EMB * 2);
  float*    ybuf = (float*)   sal((size_t)TBN * EMB * NTOK * 4);
  _Float16* qt   = (_Float16*)sal((size_t)TBN * NTOK * EMB * 2);
  unsigned char* ku = (unsigned char*)sal((size_t)TBN * EMB * NTOK);
  unsigned char* vu = (unsigned char*)sal((size_t)TBN * EMB * NTOK);
  _Float16* ktv  = (_Float16*)sal((size_t)TBN * NHEADS * 1024 * 2);
  float*    obuf = (float*)   sal((size_t)TBN * EMB * NTOK * 4);
  _Float16* oft  = (_Float16*)sal((size_t)TBN * NTOK * EMB * 2);
  _Float16* mst  = (_Float16*)sal((size_t)TBN * NTOK * EMB * 2);
  float*    mid  = (float*)   sal((size_t)TBN * 1024 * NTOK * 4);
  _Float16* midst= (_Float16*)sal((size_t)TBN * NTOK * 1024 * 2);
  float*    feat = (float*)   sal(1024 * 4);
  if (ws_size < need || scur > scratch_limit) return;

  auto packN = [&](const float* w, _Float16* o, int n) {
    pack_f16_kernel<<<(n + 255) / 256, 256, 0, stream>>>(w, o, n);
  };
  auto packC = [&](const float* w, _Float16* o, int I, int n) {
    pack_conv_kernel<<<(n + 255) / 256, 256, 0, stream>>>(w, o, I, n);
  };
  packC(F(89), w1p, 32, 18432); packC(F(90), w2p, 64, 73728);
  packC(F(91), w3p, 128, 294912); packC(F(92), w4p, 256, 589824);
  auto prep = [&](int gi, int bi, int mi, int vi, const float* bias, int slot, int C) {
    bn_prep_kernel<<<(C + 255) / 256, 256, 0, stream>>>(F(gi), F(bi), F(mi), F(vi), bias,
                                                        esA[slot], ehA[slot], C);
  };
  prep(69, 68, 70, 71, nullptr, 0, 32);
  prep(73, 72, 74, 75, nullptr, 1, 64);
  prep(77, 76, 78, 79, nullptr, 2, 128);
  prep(81, 80, 82, 83, nullptr, 3, 256);
  prep(85, 84, 86, 87, nullptr, 4, 256);
  for (int L = 0; L < 2; ++L) {
    int ab = 2 + 33 * L, sl = 5 + 6 * L;
    packN(F(ab + 15), qwp[L], 65536); packN(F(ab + 4),  kwp[L], 65536);
    packN(F(ab + 20), vwp[L], 65536); packN(F(ab + 10), pwp[L], 65536);
    packN(F(ab + 31), m1p[L], 262144); packN(F(ab + 32), m2p[L], 262144);
    prep(ab + 12, ab + 11, ab + 13, ab + 14, nullptr,     sl + 0, 256);
    prep(ab + 1,  ab + 0,  ab + 2,  ab + 3,  nullptr,     sl + 1, 256);
    prep(ab + 17, ab + 16, ab + 18, ab + 19, nullptr,     sl + 2, 256);
    prep(ab + 7,  ab + 6,  ab + 8,  ab + 9,  F(ab + 5),   sl + 3, 256);
    prep(ab + 24, ab + 23, ab + 25, ab + 26, F(ab + 21),  sl + 4, 1024);
    prep(ab + 28, ab + 27, ab + 29, ab + 30, F(ab + 22),  sl + 5, 256);
  }

  auto lif = [&](const float* X, _Float16* Sf, _Float16* Sft, unsigned char* Su,
                 float vth, long S, long xs, int C, int NT) {
    lif_kernel<<<(int)((S + 255) / 256), 256, 0, stream>>>(X, Sf, Sft, Su, vth, S, xs, C, NT);
  };
  auto gemm = [&](const _Float16* A, const _Float16* Bt, float* D, int slot,
                  const float* resid, int M, int N, int K) {
    gemm_wmma_kernel<<<dim3(N / 16, M / 64, TBN), 128, 0, stream>>>(A, Bt, D, esA[slot],
                                                                    ehA[slot], resid, M, N, K);
  };

  conv0_kernel<<<4096, 256, 0, stream>>>(xin, F(88), esA[0], ehA[0], a0);
  lif(a0, s0, nullptr, nullptr, 1.f, (long)BATCH * 32 * 16384, 0, 1, 1);
  conv3x3_wmma_kernel<<<dim3(128 * 8, 1, TBN), 128, 0, stream>>>(s0, w1p, a1, esA[1], ehA[1], 32, 64, 128, 128);
  lif(a1, s1, nullptr, nullptr, 1.f, (long)BATCH * 64 * 16384, (long)BATCH * 64 * 16384, 1, 1);
  conv3x3_wmma_kernel<<<dim3(128 * 8, 2, TBN), 128, 0, stream>>>(s1, w2p, a2, esA[2], ehA[2], 64, 128, 128, 128);
  lif(a2, s2, nullptr, nullptr, 1.f, (long)BATCH * 128 * 16384, (long)BATCH * 128 * 16384, 1, 1);
  maxpool_kernel<<<(int)(((long)TBN * 128 * 4096 + 255) / 256), 256, 0, stream>>>(
      s2, s2p, 128, 128, 128, (long)TBN * 128 * 4096);
  conv3x3_wmma_kernel<<<dim3(64 * 4, 4, TBN), 128, 0, stream>>>(s2p, w3p, a3, esA[3], ehA[3], 128, 256, 64, 64);
  lif(a3, s3, nullptr, nullptr, 1.f, (long)BATCH * 256 * 4096, (long)BATCH * 256 * 4096, 1, 1);
  maxpool_kernel<<<(int)(((long)TBN * 256 * 1024 + 255) / 256), 256, 0, stream>>>(
      s3, s3p, 256, 64, 64, (long)TBN * 256 * 1024);
  conv3x3_wmma_kernel<<<dim3(32 * 2, 4, TBN), 128, 0, stream>>>(s3p, w4p, x1, esA[4], ehA[4], 256, 256, 32, 32);
  hipMemcpyAsync(x2, x1, (size_t)TBN * EMB * NTOK * 4, hipMemcpyDeviceToDevice, stream);

  const long SA = (long)BATCH * EMB * NTOK;
  for (int L = 0; L < 2; ++L) {
    int sl = 5 + 6 * L;
    lif(x2, nullptr, st, nullptr, 1.f, SA, SA, EMB, NTOK);
    gemm(qwp[L], st, ybuf, sl + 0, nullptr, 256, 1024, 256);
    lif(ybuf, nullptr, qt, nullptr, 1.f, SA, SA, EMB, NTOK);
    gemm(kwp[L], st, ybuf, sl + 1, nullptr, 256, 1024, 256);
    lif(ybuf, nullptr, nullptr, ku, 1.f, SA, SA, EMB, NTOK);
    gemm(vwp[L], st, ybuf, sl + 2, nullptr, 256, 1024, 256);
    lif(ybuf, nullptr, nullptr, vu, 1.f, SA, SA, EMB, NTOK);
    kTv_iu8_kernel<<<TBN * NHEADS, 32, 0, stream>>>(ku, vu, ktv);
    qkTv_kernel<<<dim3(16, NHEADS, TBN), 128, 0, stream>>>(qt, ktv, obuf);
    lif(obuf, nullptr, oft, nullptr, 0.5f, SA, SA, EMB, NTOK);
    gemm(pwp[L], oft, x1, sl + 3, x1, 256, 1024, 256);
    lif(x1, nullptr, mst, nullptr, 1.f, SA, SA, EMB, NTOK);
    gemm(m1p[L], mst, mid, sl + 4, nullptr, 1024, 1024, 256);
    lif(mid, nullptr, midst, nullptr, 1.f, (long)BATCH * 1024 * NTOK,
        (long)BATCH * 1024 * NTOK, 1024, NTOK);
    gemm(m2p[L], midst, x2, sl + 5, x2, 256, 1024, 1024);
  }

  feat_kernel<<<4, 256, 0, stream>>>(x1, x2, feat);
  head_kernel<<<8, 256, 0, stream>>>(feat, hw, hb, (float*)d_out);
}
